// GeoER_56538949485200
// MI455X (gfx1250) — matrix operand order repaired and checked by reading, not compiled, not run on hardware
//
#include <hip/hip_runtime.h>
#include <hip/hip_bf16.h>
#include <math.h>

// ---------------- problem constants ----------------
constexpr int BB   = 1024;
constexpr int NN   = 64;
constexpr int HH   = 768;
constexpr int AA   = 256;
constexpr int NEMB = 256;
constexpr int CEMB = 128;
constexpr int DIN  = HH + 2 * CEMB + NEMB;  // 1280
constexpr int DH   = DIN / 2;               // 640

// ---------------- native vector types (clang ext_vector) ----------------
typedef __attribute__((ext_vector_type(16))) __bf16          v16bf;
typedef __attribute__((ext_vector_type(8)))  float           v8f;
typedef __attribute__((ext_vector_type(4)))  float           f32x4;
typedef __attribute__((ext_vector_type(4)))  unsigned short  u16x4;

union FragU { v16bf v; u16x4 q[4]; };

__device__ __forceinline__ unsigned short f2bf(float f) {
  unsigned u = __builtin_bit_cast(unsigned, f);
  u += 0x7FFFu + ((u >> 16) & 1u);      // round-to-nearest-even
  return (unsigned short)(u >> 16);
}

__device__ __forceinline__ v8f wmma_bf16(v16bf a, v16bf b, v8f c) {
  // D = A(16x32 bf16) * B(32x16 bf16) + C(16x16 f32)
  return __builtin_amdgcn_wmma_f32_16x16x32_bf16(
      /*neg_a=*/false, a, /*neg_b=*/false, b,
      /*c_mod=*/(short)0, c, /*reuse_a=*/false, /*reuse_b=*/false);
}

// A fragment (16x32, bf16) from LDS tile, row stride ld (elements).
// ISA layout: lane L -> row M = L&15; lanes<16 hold K=[0..7,16..23], lanes>=16 hold +8.
__device__ __forceinline__ v16bf frag_a(const unsigned short* Ls, int ld,
                                        int row0, int k0, int lane) {
  int r  = row0 + (lane & 15);
  int hi = (lane >> 4) * 8;
  const unsigned short* base = Ls + r * ld + k0 + hi;
  FragU f;
  f.q[0] = *(const u16x4*)(base + 0);   // K 0..3   (+hi)
  f.q[1] = *(const u16x4*)(base + 4);   // K 4..7
  f.q[2] = *(const u16x4*)(base + 16);  // K 16..19
  f.q[3] = *(const u16x4*)(base + 20);  // K 20..23
  return f.v;
}

// B fragment (32x16, bf16) built from weight tile W[n][k] (row-major, stride ld):
// B[k][n] = W[n0+n][k0+k]. lane L -> col N = L&15; lanes<16 hold K=[0..7,16..23], lanes>=16 hold +8.
__device__ __forceinline__ v16bf frag_b(const unsigned short* Ls, int ld,
                                        int n0, int k0, int lane) {
  int n  = n0 + (lane & 15);
  int kh = (lane >> 4) * 8;
  const unsigned short* base = Ls + n * ld + k0 + kh;
  FragU f;
  f.q[0] = *(const u16x4*)(base + 0);
  f.q[1] = *(const u16x4*)(base + 4);
  f.q[2] = *(const u16x4*)(base + 16);
  f.q[3] = *(const u16x4*)(base + 20);
  return f.v;
}

__device__ __forceinline__ void st_bf4(unsigned short* dst, f32x4 v) {
  u16x4 o;
  o[0] = f2bf(v[0]); o[1] = f2bf(v[1]); o[2] = f2bf(v[2]); o[3] = f2bf(v[3]);
  *(u16x4*)dst = o;
}

// =====================================================================
// Generic GEMM:  C[M,N] = act( A_f32[M,K] * W_f32[N,K]^T + bias[N] )
// block tile 64(M) x 128(N), 256 threads = 8 waves, wave does 2x2 WMMA tiles.
// Software-pipelined: next K-chunk global loads overlap current WMMA work.
// grid = (N/128, M/64)
// =====================================================================
template <int ACT>  // 0 none, 1 exact GELU
__global__ __launch_bounds__(256) void gemm_wmma(
    const float* __restrict__ Amat, int lda,
    const float* __restrict__ Wmat, int ldw,
    const float* __restrict__ bias,
    float* __restrict__ Cmat, int ldc, int Kdim) {
  constexpr int LDT = 68;  // padded bf16 row stride (136B: 8B aligned, bank-staggered)
  __shared__ __align__(16) unsigned short LA[64 * LDT];
  __shared__ __align__(16) unsigned short LW[128 * LDT];

  const int tid  = threadIdx.x;
  const int lane = tid & 31;
  const int w    = tid >> 5;
  const int m0   = blockIdx.y * 64;
  const int n0   = blockIdx.x * 128;
  const int mi2  = w & 1;   // M super-tile (covers tiles 2*mi2, 2*mi2+1)
  const int nj2  = w >> 1;  // N super-tile (covers tiles 2*nj2, 2*nj2+1)

  v8f acc[2][2] = {};
  f32x4 ra[4], rw[8];

  auto load_regs = [&](int k0) {
#pragma unroll
    for (int j = 0; j < 4; ++j) {   // A: 64x64 -> 16 floats/thread (streamed, NT)
      int i = tid + 256 * j, r = i >> 4, c4 = (i & 15) * 4;
      ra[j] = __builtin_nontemporal_load(
          (const f32x4*)(Amat + (size_t)(m0 + r) * lda + k0 + c4));
    }
#pragma unroll
    for (int j = 0; j < 8; ++j) {   // W: 128x64 (L2-resident, RT)
      int i = tid + 256 * j, r = i >> 4, c4 = (i & 15) * 4;
      rw[j] = *(const f32x4*)(Wmat + (size_t)(n0 + r) * ldw + k0 + c4);
    }
  };
  auto store_lds = [&]() {
#pragma unroll
    for (int j = 0; j < 4; ++j) {
      int i = tid + 256 * j, r = i >> 4, c4 = (i & 15) * 4;
      st_bf4(&LA[r * LDT + c4], ra[j]);
    }
#pragma unroll
    for (int j = 0; j < 8; ++j) {
      int i = tid + 256 * j, r = i >> 4, c4 = (i & 15) * 4;
      st_bf4(&LW[r * LDT + c4], rw[j]);
    }
  };

  const int nchunks = Kdim >> 6;
  load_regs(0);
  store_lds();
  __syncthreads();
  for (int c = 0; c < nchunks; ++c) {
    const bool more = (c + 1 < nchunks);
    if (more) load_regs((c + 1) << 6);   // loads in flight during WMMAs below
#pragma unroll
    for (int ks = 0; ks < 64; ks += 32) {
      v16bf af[2], bf[2];
#pragma unroll
      for (int dm = 0; dm < 2; ++dm) af[dm] = frag_a(LA, LDT, (2 * mi2 + dm) * 16, ks, lane);
#pragma unroll
      for (int dn = 0; dn < 2; ++dn) bf[dn] = frag_b(LW, LDT, (2 * nj2 + dn) * 16, ks, lane);
#pragma unroll
      for (int dm = 0; dm < 2; ++dm)
#pragma unroll
        for (int dn = 0; dn < 2; ++dn) acc[dm][dn] = wmma_bf16(af[dm], bf[dn], acc[dm][dn]);
    }
    __syncthreads();
    if (more) {
      store_lds();
      __syncthreads();
    }
  }

  // epilogue: C layout lane L -> N = L&15, VGPR v -> M = v + 8*(L>=16)
  const int nl = lane & 15;
  const int mh = (lane >> 4) * 8;
#pragma unroll
  for (int dm = 0; dm < 2; ++dm) {
#pragma unroll
    for (int dn = 0; dn < 2; ++dn) {
      int col  = n0 + (2 * nj2 + dn) * 16 + nl;
      float bs = bias ? bias[col] : 0.f;
#pragma unroll
      for (int v = 0; v < 8; ++v) {
        int row = m0 + (2 * mi2 + dm) * 16 + mh + v;
        float x = acc[dm][dn][v] + bs;
        if (ACT == 1) x = 0.5f * x * (1.0f + erff(x * 0.70710678f));
        Cmat[(size_t)row * ldc + col] = x;
      }
    }
  }
}

// =====================================================================
// Fused neighbor pool: one block per batch element b.
// wnei[64,256] = neigh[b] @ W^T + bias  (WMMA, K=768 streamed through LDS,
// software-pipelined), then score/leaky/dist-bias/mask/softmax/context/sim/ReLU.
// =====================================================================
__global__ __launch_bounds__(256) void pool_wmma(
    const float* __restrict__ neigh,      // [B,N,H]
    const float* __restrict__ wattn,      // [A,H]
    const float* __restrict__ wattn_b,    // [A]
    const float* __restrict__ wn,         // [B,A] precomputed node @ W^T + b
    const float* __restrict__ attn_w,     // [2A]
    const float* __restrict__ attn_b,     // [1]
    const float* __restrict__ b_attn_w,   // [1]
    const float* __restrict__ b_attn_b,   // [1]
    const float* __restrict__ dist,       // [B,N]
    const unsigned char* __restrict__ mask,  // [B,N] bool
    float* __restrict__ simout,           // [B, 2A]
    int col_off) {
  constexpr int LDT = 68;
  constexpr int LDW = 257;  // f32 wnei row stride (odd -> conflict-free)
  __shared__ __align__(16) unsigned short LA[64 * LDT];    // neigh chunk (bf16)
  __shared__ __align__(16) unsigned short LW[256 * LDT];   // weight chunk (bf16)
  __shared__ __align__(16) float WN[64 * LDW];             // wnei (f32)
  __shared__ float sc[64];
  __shared__ float esc[64];
  __shared__ float red[8];

  const int tid  = threadIdx.x;
  const int lane = tid & 31;
  const int w    = tid >> 5;
  const int b    = blockIdx.x;
  const int mi2  = w & 1;   // M tiles 2*mi2, 2*mi2+1
  const int nj4  = w >> 1;  // N tiles 4*nj4 .. 4*nj4+3

  const float* nb = neigh + (size_t)b * NN * HH;
  v8f acc[2][4] = {};
  f32x4 ra[4], rw[16];

  auto load_regs = [&](int k0) {
#pragma unroll
    for (int j = 0; j < 4; ++j) {   // neigh chunk: read-once stream -> NT
      int i = tid + 256 * j, r = i >> 4, c4 = (i & 15) * 4;
      ra[j] = __builtin_nontemporal_load(
          (const f32x4*)(nb + (size_t)r * HH + k0 + c4));
    }
#pragma unroll
    for (int j = 0; j < 16; ++j) {  // weights: L2-resident
      int i = tid + 256 * j, r = i >> 4, c4 = (i & 15) * 4;
      rw[j] = *(const f32x4*)(wattn + (size_t)r * HH + k0 + c4);
    }
  };
  auto store_lds = [&]() {
#pragma unroll
    for (int j = 0; j < 4; ++j) {
      int i = tid + 256 * j, r = i >> 4, c4 = (i & 15) * 4;
      st_bf4(&LA[r * LDT + c4], ra[j]);
    }
#pragma unroll
    for (int j = 0; j < 16; ++j) {
      int i = tid + 256 * j, r = i >> 4, c4 = (i & 15) * 4;
      st_bf4(&LW[r * LDT + c4], rw[j]);
    }
  };

  constexpr int NCH = HH >> 6;  // 12
  load_regs(0);
  store_lds();
  __syncthreads();
  for (int c = 0; c < NCH; ++c) {
    const bool more = (c + 1 < NCH);
    if (more) load_regs((c + 1) << 6);   // HBM stream overlaps WMMAs
#pragma unroll
    for (int ks = 0; ks < 64; ks += 32) {
      v16bf af[2], bf[4];
#pragma unroll
      for (int dm = 0; dm < 2; ++dm) af[dm] = frag_a(LA, LDT, (2 * mi2 + dm) * 16, ks, lane);
#pragma unroll
      for (int dn = 0; dn < 4; ++dn) bf[dn] = frag_b(LW, LDT, (4 * nj4 + dn) * 16, ks, lane);
#pragma unroll
      for (int dm = 0; dm < 2; ++dm)
#pragma unroll
        for (int dn = 0; dn < 4; ++dn) acc[dm][dn] = wmma_bf16(af[dm], bf[dn], acc[dm][dn]);
    }
    __syncthreads();
    if (more) {
      store_lds();
      __syncthreads();
    }
  }

  // park wnei (+bias) in LDS
  const int nl = lane & 15;
  const int mh = (lane >> 4) * 8;
#pragma unroll
  for (int dm = 0; dm < 2; ++dm)
#pragma unroll
    for (int dn = 0; dn < 4; ++dn) {
      int col  = (4 * nj4 + dn) * 16 + nl;
      float bs = wattn_b[col];
#pragma unroll
      for (int v = 0; v < 8; ++v) {
        int row = (2 * mi2 + dm) * 16 + mh + v;
        WN[row * LDW + col] = acc[dm][dn][v] + bs;
      }
    }

  // s0 = dot(wn[b], attn_w[0:A])  (wave shuffle + cross-wave reduce)
  const float* wnb = wn + (size_t)b * AA;
  float p = wnb[tid] * attn_w[tid];
#pragma unroll
  for (int m = 16; m; m >>= 1) p += __shfl_xor(p, m, 32);
  if (lane == 0) red[w] = p;
  __syncthreads();
  float s0 = red[0] + red[1] + red[2] + red[3] + red[4] + red[5] + red[6] + red[7];

  // scores, leaky, distance bias, mask
  if (tid < NN) {
    float a = 0.f;
    for (int j = 0; j < AA; ++j) a += WN[tid * LDW + j] * attn_w[AA + j];
    float s = s0 + a + attn_b[0];
    s = (s >= 0.f) ? s : 0.01f * s;
    s += dist[(size_t)b * NN + tid] * b_attn_w[0] + b_attn_b[0];
    if (!mask[(size_t)b * NN + tid]) s = -1e30f;
    sc[tid] = s;
  }
  __syncthreads();
  if (tid < NN) {
    float mx = sc[0];
    for (int n = 1; n < NN; ++n) mx = fmaxf(mx, sc[n]);
    esc[tid] = __expf(sc[tid] - mx);
  }
  __syncthreads();
  if (tid < NN) {
    float sum = 0.f;
    for (int n = 0; n < NN; ++n) sum += esc[n];
    sc[tid] = esc[tid] / sum;  // att
  }
  __syncthreads();

  // context (thread t owns feature a=t), sim, ReLU
  {
    float ctx = 0.f;
    for (int n = 0; n < NN; ++n) ctx += sc[n] * WN[n * LDW + tid];
    float sim = ctx * wnb[tid];
    sim = sim > 0.f ? sim : 0.f;
    simout[(size_t)b * (2 * AA) + col_off + tid] = sim;
  }
}

// ---------------- tail kernels ----------------
__global__ __launch_bounds__(256) void assemble_xcat(
    const float* __restrict__ pooled, const float* __restrict__ x_coord,
    const float* __restrict__ coord_w, const float* __restrict__ coord_b,
    const float* __restrict__ xnb, float* __restrict__ xcat) {
  int b = blockIdx.x, t = threadIdx.x;
  float* row = xcat + (size_t)b * DIN;
  for (int i = t; i < HH; i += 256) row[i] = pooled[(size_t)b * HH + i];
  float xc = x_coord[b];
  for (int i = t; i < 2 * CEMB; i += 256) row[HH + i] = xc * coord_w[i] + coord_b[i];
  for (int i = t; i < NEMB; i += 256) row[HH + 2 * CEMB + i] = xnb[(size_t)b * NEMB + i];
}

__global__ __launch_bounds__(256) void head_kernel(
    const float* __restrict__ h, const float* __restrict__ w2,
    const float* __restrict__ b2, float* __restrict__ out) {
  int b = blockIdx.x * blockDim.x + threadIdx.x;
  if (b >= BB) return;
  const float* hr = h + (size_t)b * DH;
  float l0 = b2[0], l1 = b2[1];
  for (int k = 0; k < DH; ++k) {
    float x = hr[k];
    l0 += x * w2[k];
    l1 += x * w2[DH + k];
  }
  float mx  = fmaxf(l0, l1);
  float lse = mx + logf(expf(l0 - mx) + expf(l1 - mx));
  out[2 * b + 0] = l0 - lse;
  out[2 * b + 1] = l1 - lse;
}

// ---------------- launch ----------------
extern "C" void kernel_launch(void* const* d_in, const int* in_sizes, int n_in,
                              void* d_out, int out_size, void* d_ws, size_t ws_size,
                              hipStream_t stream) {
  (void)in_sizes; (void)n_in; (void)out_size; (void)ws_size;
  const float* pooled   = (const float*)d_in[0];
  const float* node1    = (const float*)d_in[1];
  const float* node2    = (const float*)d_in[2];
  const float* neigh1   = (const float*)d_in[3];
  const float* neigh2   = (const float*)d_in[4];
  const float* dist1    = (const float*)d_in[5];
  const float* dist2    = (const float*)d_in[6];
  const unsigned char* mask1 = (const unsigned char*)d_in[7];
  const unsigned char* mask2 = (const unsigned char*)d_in[8];
  const float* x_coord  = (const float*)d_in[9];
  const float* w_attn_w = (const float*)d_in[10];
  const float* w_attn_b = (const float*)d_in[11];
  const float* attn_w   = (const float*)d_in[12];
  const float* attn_b   = (const float*)d_in[13];
  const float* b_attn_w = (const float*)d_in[14];
  const float* b_attn_b = (const float*)d_in[15];
  const float* neigh_w  = (const float*)d_in[16];
  const float* neigh_b  = (const float*)d_in[17];
  const float* coord_w  = (const float*)d_in[18];
  const float* coord_b  = (const float*)d_in[19];
  const float* lin1_w   = (const float*)d_in[20];
  const float* lin1_b   = (const float*)d_in[21];
  const float* lin2_w   = (const float*)d_in[22];
  const float* lin2_b   = (const float*)d_in[23];
  float* out = (float*)d_out;

  float* ws   = (float*)d_ws;
  float* wn1  = ws;                  ws += (size_t)BB * AA;
  float* wn2  = ws;                  ws += (size_t)BB * AA;
  float* sim  = ws;                  ws += (size_t)BB * 2 * AA;
  float* xnb  = ws;                  ws += (size_t)BB * NEMB;
  float* xcat = ws;                  ws += (size_t)BB * DIN;
  float* hbuf = ws;                  ws += (size_t)BB * DH;

  dim3 blk(256);
  // wn1/wn2 = node @ W^T + b   (M=1024, N=256, K=768)
  gemm_wmma<0><<<dim3(AA / 128, BB / 64), blk, 0, stream>>>(node1, HH, w_attn_w, HH,
                                                            w_attn_b, wn1, AA, HH);
  gemm_wmma<0><<<dim3(AA / 128, BB / 64), blk, 0, stream>>>(node2, HH, w_attn_w, HH,
                                                            w_attn_b, wn2, AA, HH);
  // fused neighbor pools
  pool_wmma<<<BB, blk, 0, stream>>>(neigh1, w_attn_w, w_attn_b, wn1, attn_w, attn_b,
                                    b_attn_w, b_attn_b, dist1, mask1, sim, 0);
  pool_wmma<<<BB, blk, 0, stream>>>(neigh2, w_attn_w, w_attn_b, wn2, attn_w, attn_b,
                                    b_attn_w, b_attn_b, dist2, mask2, sim, AA);
  // x_neighbors = relu(sim) @ neigh_w^T + b   (M=1024, N=256, K=512)
  gemm_wmma<0><<<dim3(NEMB / 128, BB / 64), blk, 0, stream>>>(sim, 2 * AA, neigh_w, 2 * AA,
                                                              neigh_b, xnb, NEMB, 2 * AA);
  // concat [pooled | coord | x_neighbors]
  assemble_xcat<<<BB, blk, 0, stream>>>(pooled, x_coord, coord_w, coord_b, xnb, xcat);
  // h = gelu(xcat @ lin1_w^T + b)   (M=1024, N=640, K=1280)
  gemm_wmma<1><<<dim3(DH / 128, BB / 64), blk, 0, stream>>>(xcat, DIN, lin1_w, DIN,
                                                            lin1_b, hbuf, DH, DIN);
  // logits + log_softmax
  head_kernel<<<BB / 256, blk, 0, stream>>>(hbuf, lin2_w, lin2_b, out);
}
